// MultiHeadAttention_80814104641872
// MI455X (gfx1250) — matrix operand order repaired and checked by reading, not compiled, not run on hardware
//
#include <hip/hip_runtime.h>
#include <hip/hip_bf16.h>

typedef _Float16 half_t;
typedef __attribute__((ext_vector_type(4)))  float    v4f;
typedef __attribute__((ext_vector_type(8)))  float    v8f;
typedef __attribute__((ext_vector_type(8)))  _Float16 v8h;
typedef __attribute__((ext_vector_type(16))) _Float16 v16h;

#define HIDDEN 1024
#define HEADS 16
#define HEAD_DIM 64
#define BATCH 2
#define LQ 2048
#define LK 2048
#define MTOT (BATCH * LQ)   // 4096

static __device__ __forceinline__ v16h combine8(v8h lo, v8h hi) {
    return __builtin_shufflevector(lo, hi, 0,1,2,3,4,5,6,7,8,9,10,11,12,13,14,15);
}

static __device__ __forceinline__ v8f wmma_f16(v16h a, v16h b, v8f c) {
    // D = A(16x32) * B(32x16) + C, fp32 accumulate
    return __builtin_amdgcn_wmma_f32_16x16x32_f16(false, a, false, b, (short)0, c, false, false);
}

// ---------------------------------------------------------------------------
// Weight prep: fp32 (K x N) row-major -> f16 transposed (N x K) row-major
// ---------------------------------------------------------------------------
__global__ __launch_bounds__(256) void weight_transpose_f16_kernel(
        const float* __restrict__ W, half_t* __restrict__ WT) {
    int idx = blockIdx.x * blockDim.x + threadIdx.x;   // 0 .. 1M-1
    int k = idx >> 10;
    int n = idx & 1023;
    WT[(size_t)n * HIDDEN + k] = (half_t)W[(size_t)k * HIDDEN + n];
}

// ---------------------------------------------------------------------------
// LDS staging helpers (16 contiguous elements per thread, vectorized stores)
// ---------------------------------------------------------------------------
static __device__ __forceinline__ void stage_a16(const float* __restrict__ p,
                                                 half_t* __restrict__ d) {
    const v4f* pv = (const v4f*)p;
    v4f f0 = pv[0], f1 = pv[1], f2 = pv[2], f3 = pv[3];
    v8h h0, h1;
#pragma unroll
    for (int i = 0; i < 4; ++i) {
        h0[i]     = (half_t)f0[i];
        h0[i + 4] = (half_t)f1[i];
        h1[i]     = (half_t)f2[i];
        h1[i + 4] = (half_t)f3[i];
    }
    *(v8h*)d       = h0;
    *(v8h*)(d + 8) = h1;
}
static __device__ __forceinline__ void stage_a16(const half_t* __restrict__ p,
                                                 half_t* __restrict__ d) {
    *(v8h*)d       = *(const v8h*)p;
    *(v8h*)(d + 8) = *(const v8h*)(p + 8);
}

// ---------------------------------------------------------------------------
// Tiled WMMA GEMM:  C(M=4096, N=1024) = sscale * A(M,1024) @ B(1024,1024)
//   A: row-major (fp32 or f16), BT: f16 B-transposed (N x K) row-major
//   MODE 0: store f16 into (bh, l, d) head-split layout   (Q, K)
//   MODE 1: store f16 into (bh, d, l) transposed layout   (V)
//   MODE 2: store fp32 row-major into d_out               (output proj)
// Block: 256 threads (8 waves), tile 128x64, BK=32.
// ---------------------------------------------------------------------------
template <typename AT, int MODE>
__global__ __launch_bounds__(256) void gemm_wmma_kernel(
        const AT* __restrict__ A, const half_t* __restrict__ BT,
        void* __restrict__ Cout, float sscale) {
    constexpr int LDS_STRIDE = 40;  // 32 + 8 halves pad (80B rows: conflict-free b128)
    __shared__ __align__(16) half_t As[128 * LDS_STRIDE];
    __shared__ __align__(16) half_t Bs[64 * LDS_STRIDE];

    const int tid  = threadIdx.x;
    const int wave = tid >> 5;
    const int lane = tid & 31;
    const int lr   = lane & 15;
    const int hi   = lane >> 4;

    const int mbase = blockIdx.x * 128;
    const int nbase = blockIdx.y * 64;

    const v8f vzero = {0.f, 0.f, 0.f, 0.f, 0.f, 0.f, 0.f, 0.f};
    v8f acc[4] = {vzero, vzero, vzero, vzero};

    const int ar = tid >> 1;            // 0..127
    const int ac = (tid & 1) * 16;      // 0 or 16
    const int br = tid >> 2;            // 0..63
    const int bc = (tid & 3) * 8;       // 0,8,16,24

    for (int kt = 0; kt < HIDDEN / 32; ++kt) {
        const int kbase = kt * 32;
        // stage A tile 128x32 (convert fp32->f16 in flight if needed)
        stage_a16(A + (size_t)(mbase + ar) * HIDDEN + kbase + ac,
                  &As[ar * LDS_STRIDE + ac]);
        // stage BT tile 64x32 (already f16, already transposed)
        *(v8h*)&Bs[br * LDS_STRIDE + bc] =
            *(const v8h*)(BT + (size_t)(nbase + br) * HIDDEN + kbase + bc);
        __syncthreads();

        // A fragment: lane = row (wave*16 + lr), k = e + 16*hi
        const int arow = wave * 16 + lr;
        v16h afrag = combine8(*(const v8h*)&As[arow * LDS_STRIDE + 16 * hi],
                              *(const v8h*)&As[arow * LDS_STRIDE + 16 * hi + 8]);
#pragma unroll
        for (int j = 0; j < 4; ++j) {
            const int c = j * 16 + lr;  // lane = output column, k = e + 16*hi
            v16h bfrag = combine8(*(const v8h*)&Bs[c * LDS_STRIDE + 16 * hi],
                                  *(const v8h*)&Bs[c * LDS_STRIDE + 16 * hi + 8]);
            acc[j] = wmma_f16(afrag, bfrag, acc[j]);
        }
        __syncthreads();
    }

    // D layout: VGPR v holds row (v + 8*hi), col = lr within each 16-col frag
    const int m0 = mbase + wave * 16;
#pragma unroll
    for (int j = 0; j < 4; ++j) {
#pragma unroll
        for (int v = 0; v < 8; ++v) {
            const int m = m0 + v + 8 * hi;
            const int n = nbase + j * 16 + lr;
            const float val = acc[j][v] * sscale;
            if (MODE == 2) {
                ((float*)Cout)[(size_t)m * HIDDEN + n] = val;
            } else {
                const int b = m >> 11;       // m / LQ
                const int l = m & (LQ - 1);
                const int h = n >> 6;        // n / HEAD_DIM
                const int d = n & (HEAD_DIM - 1);
                const size_t bh = (size_t)b * HEADS + h;
                if (MODE == 0) {  // (bh, l, d)
                    ((half_t*)Cout)[(bh * LK + l) * HEAD_DIM + d] = (half_t)val;
                } else {          // (bh, d, l)
                    ((half_t*)Cout)[(bh * HEAD_DIM + d) * LK + l] = (half_t)val;
                }
            }
        }
    }
}

// ---------------------------------------------------------------------------
// Flash attention per (bh, 128-row q tile). 8 waves x 16 q rows.
// Qh (pre-scaled by 1/sqrt(d)) / Kh: (bh, l, d) f16.  VhT: (bh, d, l) f16.
// O16: (b, l, HIDDEN) f16.
// ---------------------------------------------------------------------------
__global__ __launch_bounds__(256) void attn_flash_kernel(
        const half_t* __restrict__ Qh, const half_t* __restrict__ Kh,
        const half_t* __restrict__ VhT, const int* __restrict__ valid_lens,
        half_t* __restrict__ O16) {
    constexpr int PSTRIDE = 72;  // 64 + 8 halves pad
    __shared__ __align__(16) half_t Plds[8 * 16 * PSTRIDE];

    const int bh   = blockIdx.y;           // 0..31
    const int b    = bh >> 4;
    const int head = bh & (HEADS - 1);
    int vl = valid_lens[b];
    if (vl < 0) vl = 0;
    if (vl > LK) vl = LK;

    const int tid  = threadIdx.x;
    const int wave = tid >> 5;
    const int lane = tid & 31;
    const int lr   = lane & 15;
    const int hi   = lane >> 4;
    const float NEG_INF = -__builtin_inff();

    // Q A-fragments (already scaled by 1/sqrt(d)), resident for the whole loop
    const int qrow = blockIdx.x * 128 + wave * 16 + lr;
    const half_t* qp = Qh + ((size_t)bh * LQ + qrow) * HEAD_DIM;
    v16h qfrag[2];
#pragma unroll
    for (int t = 0; t < 2; ++t)
        qfrag[t] = combine8(*(const v8h*)(qp + 32 * t + 16 * hi),
                            *(const v8h*)(qp + 32 * t + 16 * hi + 8));

    const v8f vzero = {0.f, 0.f, 0.f, 0.f, 0.f, 0.f, 0.f, 0.f};
    v8f oacc[4] = {vzero, vzero, vzero, vzero};
    float mrun[8], lrun[8];
#pragma unroll
    for (int v = 0; v < 8; ++v) { mrun[v] = NEG_INF; lrun[v] = 0.f; }

    half_t* pw = &Plds[wave * 16 * PSTRIDE];  // per-wave private patch
    const int kblocks = (vl + 63) >> 6;

    for (int kb = 0; kb < kblocks; ++kb) {
        const int k0 = kb * 64;

        // --- S = (Q/sqrt(d)) @ K^T, 16 x 64 per wave ---
        v8f s[4] = {vzero, vzero, vzero, vzero};
        bool valid[4];
#pragma unroll
        for (int j = 0; j < 4; ++j) {
            const int kc = k0 + j * 16 + lr;               // key column
            valid[j] = (kc < vl);
            const half_t* kp = Kh + ((size_t)bh * LK + kc) * HEAD_DIM;
            v16h b0 = combine8(*(const v8h*)(kp + 16 * hi),
                               *(const v8h*)(kp + 16 * hi + 8));
            v16h b1 = combine8(*(const v8h*)(kp + 32 + 16 * hi),
                               *(const v8h*)(kp + 32 + 16 * hi + 8));
            s[j] = wmma_f16(qfrag[0], b0, s[j]);
            s[j] = wmma_f16(qfrag[1], b1, s[j]);
        }

        // --- online softmax (rows v + 8*hi; 16-lane half-wave reductions) ---
        float rmax[8];
#pragma unroll
        for (int v = 0; v < 8; ++v) {
            float x0 = valid[0] ? s[0][v] : NEG_INF;
            float x1 = valid[1] ? s[1][v] : NEG_INF;
            float x2 = valid[2] ? s[2][v] : NEG_INF;
            float x3 = valid[3] ? s[3][v] : NEG_INF;
            rmax[v] = fmaxf(fmaxf(x0, x1), fmaxf(x2, x3));
        }
#pragma unroll
        for (int v = 0; v < 8; ++v) {
#pragma unroll
            for (int off = 1; off < 16; off <<= 1)
                rmax[v] = fmaxf(rmax[v], __shfl_xor(rmax[v], off, 32));
        }
#pragma unroll
        for (int v = 0; v < 8; ++v) {
            const float mnew = fmaxf(mrun[v], rmax[v]);
            const float alpha = (mnew == NEG_INF) ? 1.0f : __expf(mrun[v] - mnew);
            mrun[v] = mnew;
            lrun[v] *= alpha;
#pragma unroll
            for (int jd = 0; jd < 4; ++jd) oacc[jd][v] *= alpha;
            float psum = 0.f;
#pragma unroll
            for (int j = 0; j < 4; ++j) {
                const float p = valid[j] ? __expf(s[j][v] - mnew) : 0.f;
                psum += p;
                pw[(v + 8 * hi) * PSTRIDE + j * 16 + lr] = (half_t)p;
            }
            lrun[v] += psum;
        }
        // No barrier needed: the P patch is wave-private, LDS ops are in-order
        // per wave (DScnt) and the wave executes in lockstep.

        // --- P A-fragments from LDS, then O += P @ V ---
        v16h pfrag[2];
#pragma unroll
        for (int t = 0; t < 2; ++t)
            pfrag[t] = combine8(*(const v8h*)&pw[lr * PSTRIDE + 32 * t + 16 * hi],
                                *(const v8h*)&pw[lr * PSTRIDE + 32 * t + 16 * hi + 8]);
#pragma unroll
        for (int jd = 0; jd < 4; ++jd) {
            const int d = jd * 16 + lr;
            const half_t* vp = VhT + ((size_t)bh * HEAD_DIM + d) * LK + k0;
            v16h v0 = combine8(*(const v8h*)(vp + 16 * hi),
                               *(const v8h*)(vp + 16 * hi + 8));
            v16h v1 = combine8(*(const v8h*)(vp + 32 + 16 * hi),
                               *(const v8h*)(vp + 32 + 16 * hi + 8));
            oacc[jd] = wmma_f16(pfrag[0], v0, oacc[jd]);
            oacc[jd] = wmma_f16(pfrag[1], v1, oacc[jd]);
        }
    }

    // row-sum reduce l across each 16-lane half
#pragma unroll
    for (int v = 0; v < 8; ++v) {
#pragma unroll
        for (int off = 1; off < 16; off <<= 1)
            lrun[v] += __shfl_xor(lrun[v], off, 32);
    }

    // store: D layout rows v + 8*hi
    const int q0 = blockIdx.x * 128 + wave * 16;
#pragma unroll
    for (int jd = 0; jd < 4; ++jd) {
#pragma unroll
        for (int v = 0; v < 8; ++v) {
            const float l = lrun[v];
            const float val = (l > 0.f) ? oacc[jd][v] / l : 0.f;
            const int qr = q0 + v + 8 * hi;
            O16[((size_t)b * LQ + qr) * HIDDEN + head * HEAD_DIM + jd * 16 + lr] =
                (half_t)val;
        }
    }
}

// ---------------------------------------------------------------------------
// Launch
// ---------------------------------------------------------------------------
extern "C" void kernel_launch(void* const* d_in, const int* in_sizes, int n_in,
                              void* d_out, int out_size, void* d_ws, size_t ws_size,
                              hipStream_t stream) {
    const float* q    = (const float*)d_in[0];
    const float* k    = (const float*)d_in[1];
    const float* v    = (const float*)d_in[2];
    const int*   vlen = (const int*)d_in[3];
    const float* Wq   = (const float*)d_in[4];
    const float* Wk   = (const float*)d_in[5];
    const float* Wv   = (const float*)d_in[6];
    const float* Wout = (const float*)d_in[7];

    const size_t MB1 = (size_t)HIDDEN * HIDDEN;       // 1,048,576 halves
    const size_t SZX = (size_t)MTOT * HIDDEN;         // 4,194,304 halves
    half_t* wsH = (half_t*)d_ws;
    half_t* WQT = wsH + 0 * MB1;
    half_t* WKT = wsH + 1 * MB1;
    half_t* WVT = wsH + 2 * MB1;
    half_t* WOT = wsH + 3 * MB1;
    half_t* QH  = wsH + 4 * MB1;
    half_t* KH  = wsH + 4 * MB1 + 1 * SZX;
    half_t* VHT = wsH + 4 * MB1 + 2 * SZX;
    half_t* O16 = wsH + 4 * MB1 + 3 * SZX;

    // 1) weights -> transposed f16
    weight_transpose_f16_kernel<<<MB1 / 256, 256, 0, stream>>>(Wq, WQT);
    weight_transpose_f16_kernel<<<MB1 / 256, 256, 0, stream>>>(Wk, WKT);
    weight_transpose_f16_kernel<<<MB1 / 256, 256, 0, stream>>>(Wv, WVT);
    weight_transpose_f16_kernel<<<MB1 / 256, 256, 0, stream>>>(Wout, WOT);

    // 2) projections (WMMA GEMMs); Q pre-scaled by 1/sqrt(HEAD_DIM) (exact in f16)
    dim3 ggrid(MTOT / 128, HIDDEN / 64);  // (32, 16)
    gemm_wmma_kernel<float, 0><<<ggrid, 256, 0, stream>>>(q, WQT, QH, 0.125f);
    gemm_wmma_kernel<float, 0><<<ggrid, 256, 0, stream>>>(k, WKT, KH, 1.0f);
    gemm_wmma_kernel<float, 1><<<ggrid, 256, 0, stream>>>(v, WVT, VHT, 1.0f);

    // 3) flash attention with length mask
    dim3 agrid(LQ / 128, BATCH * HEADS);  // (16, 32)
    attn_flash_kernel<<<agrid, 256, 0, stream>>>(QH, KH, VHT, vlen, O16);

    // 4) output projection -> fp32 d_out
    gemm_wmma_kernel<half_t, 2><<<ggrid, 256, 0, stream>>>(O16, WOT, (float*)d_out, 1.0f);
}
